// SparseUNetMultiTask_8065948582436
// MI455X (gfx1250) — matrix operand order repaired and checked by reading, not compiled, not run on hardware
//
#include <hip/hip_runtime.h>
#include <math.h>

#define M_FINE   64000
#define N_COARSE 16000
#define KNBR     27

typedef __attribute__((ext_vector_type(16))) __bf16 v16bf;
typedef __attribute__((ext_vector_type(8)))  float  v8f;

union BFrag {
  v16bf v;
  unsigned short s[16];
  uint4 q[2];
};

__device__ __forceinline__ unsigned short f2bf(float f) {
  union { float f; unsigned u; } x; x.f = f;
  unsigned r = x.u + 0x7FFFu + ((x.u >> 16) & 1u);   // round-to-nearest-even
  return (unsigned short)(r >> 16);
}

// ---------------------------------------------------------------------------
// Weight packing: f32 W[Kk][Cin][CoutSrc] -> bf16 fragments in CONSUMPTION
// order [Kk][NT][KB][512]: fragment f = (k*NT + nt)*KB + kb, element index =
// f*512 + lane*16 + i, with (k_local = 32*kb + 8*(lane/16) + 16*(i/8) + i%8,
// n = nt*16 + lane%16). Cout zero-padded to NT*16.
// ---------------------------------------------------------------------------
__global__ void __launch_bounds__(256)
pack_weights_kernel(const float* __restrict__ W,
                    unsigned short* __restrict__ Wp,
                    int Kk, int Cin, int CoutSrc, int NT) {
  int KB = Cin >> 5;
  int total = Kk * KB * NT * 512;
  int idx = blockIdx.x * blockDim.x + threadIdx.x;
  if (idx >= total) return;
  int e    = idx & 15;
  int lane = (idx >> 4) & 31;
  int f    = idx >> 9;
  int kb   = f % KB;
  int nt   = (f / KB) % NT;
  int k    = f / (KB * NT);
  int kk   = 32 * kb + 8 * (lane >> 4) + 16 * (e >> 3) + (e & 7);
  int n    = nt * 16 + (lane & 15);
  float v  = (n < CoutSrc) ? W[((long)k * Cin + kk) * (long)CoutSrc + n] : 0.0f;
  Wp[idx] = f2bf(v);
}

// f32 -> bf16 elementwise copy
__global__ void __launch_bounds__(256)
cvt_bf16_kernel(const float* __restrict__ src,
                unsigned short* __restrict__ dst, int total) {
  int i = blockIdx.x * blockDim.x + threadIdx.x;
  if (i < total) dst[i] = f2bf(src[i]);
}

// Branchless masked A-fragment load from a 64-channel bf16 row.
// A layout: element i -> K = 32*kb + 16*(i/8) + 8*(lane/16) + i%8,
// i.e. per fragment two contiguous 16B chunks; chunk uint4 index = ch/8.
__device__ __forceinline__ void loadA64(BFrag a[2], const unsigned short* __restrict__ Xbf,
                                        int rbv, int h) {
  int rrow = (rbv >= 0) ? rbv : 0;
  const uint4* p = (const uint4*)Xbf + (unsigned)rrow * 8u;   // 8 uint4 per row
  uint4 z = make_uint4(0u, 0u, 0u, 0u);
  uint4 t0 = p[h], t1 = p[2 + h], t2 = p[4 + h], t3 = p[6 + h];
  bool ok = rbv >= 0;
  a[0].q[0] = ok ? t0 : z;
  a[0].q[1] = ok ? t1 : z;
  a[1].q[0] = ok ? t2 : z;
  a[1].q[1] = ok ? t3 : z;
}

// Load both kb fragments of one nt column-tile (frag base f = (k*NT+nt)*2).
__device__ __forceinline__ void loadB2(BFrag bb[2], const uint4* __restrict__ wq,
                                       int fbase, int lane) {
  unsigned o = (unsigned)fbase * 64u + (unsigned)lane * 2u;
  bb[0].q[0] = wq[o];
  bb[0].q[1] = wq[o + 1];
  bb[1].q[0] = wq[o + 64];
  bb[1].q[1] = wq[o + 65];
}

// ---------------------------------------------------------------------------
// Core: masked sparse-conv accumulation, Cin=64. MT row-tiles of 16 rows
// share every B fragment. B is double-buffered (ping-pong) with cross-k
// prefetch; next-k rulebook/A gathers issue under the WMMA chain.
// cnt[mt] = valid-neighbor count for row (rowBase + mt*16 + lane%16).
// ---------------------------------------------------------------------------
template <int NT, int MT>
__device__ __forceinline__ void skconv_accum(const unsigned short* __restrict__ Xbf,
                                             const int* __restrict__ rb,
                                             const unsigned short* __restrict__ Wp,
                                             int rowBase, v8f (&c)[MT][NT], int (&cnt)[MT]) {
  static_assert((NT & 1) == 0, "NT must be even for ping-pong parity");
  const int lane = threadIdx.x & 31;
  const int m = lane & 15, h = lane >> 4;
  const uint4* wq = (const uint4*)Wp;
#pragma unroll
  for (int mt = 0; mt < MT; ++mt) {
    cnt[mt] = 0;
#pragma unroll
    for (int nt = 0; nt < NT; ++nt) {
      v8f z = {0.f, 0.f, 0.f, 0.f, 0.f, 0.f, 0.f, 0.f};
      c[mt][nt] = z;
    }
  }
  int rbv[MT];
  BFrag a[MT][2];
#pragma unroll
  for (int mt = 0; mt < MT; ++mt) {
    rbv[mt] = rb[(rowBase + mt * 16 + m) * KNBR + 0];
    loadA64(a[mt], Xbf, rbv[mt], h);
  }
  BFrag b[2][2];                      // ping-pong B buffers
  loadB2(b[0], wq, 0, lane);          // k=0, nt=0
  for (int k = 0; k < KNBR; ++k) {
    int rbn[MT];
    bool more = (k + 1) < KNBR;
#pragma unroll
    for (int mt = 0; mt < MT; ++mt) {
      rbn[mt] = more ? rb[(rowBase + mt * 16 + m) * KNBR + k + 1] : -1;
      cnt[mt] += (rbv[mt] >= 0) ? 1 : 0;
    }
#pragma unroll
    for (int nt = 0; nt < NT; ++nt) {
      const int cb = nt & 1, pb = cb ^ 1;
      if (nt + 1 < NT)
        loadB2(b[pb], wq, (k * NT + nt + 1) * 2, lane);      // next nt
      else if (more)
        loadB2(b[pb], wq, ((k + 1) * NT) * 2, lane);         // next k, nt=0
#pragma unroll
      for (int mt = 0; mt < MT; ++mt) {
        c[mt][nt] = __builtin_amdgcn_wmma_f32_16x16x32_bf16(
            false, a[mt][0].v, false, b[cb][0].v, (short)0, c[mt][nt], false, false);
        c[mt][nt] = __builtin_amdgcn_wmma_f32_16x16x32_bf16(
            false, a[mt][1].v, false, b[cb][1].v, (short)0, c[mt][nt], false, false);
      }
    }
    if (more) {
#pragma unroll
      for (int mt = 0; mt < MT; ++mt) {
        rbv[mt] = rbn[mt];
        loadA64(a[mt], Xbf, rbv[mt], h);     // overlaps WMMA execution
      }
    }
  }
}

// conv1: x1 = skconv(feats, rb1, w1) + b1 + feats        (Cout=64, 32 rows/wave)
__global__ void __launch_bounds__(256)
conv1_kernel(const unsigned short* __restrict__ featsbf,
             const float* __restrict__ feats,
             const int* __restrict__ rb1,
             const unsigned short* __restrict__ w1p,
             const float* __restrict__ b1,
             float* __restrict__ x1, int nRows) {
  int wave = blockIdx.x * (blockDim.x >> 5) + (threadIdx.x >> 5);
  int rowBase = wave * 32;
  if (rowBase >= nRows) return;
  v8f c[2][4];
  int cnt[2];
  skconv_accum<4, 2>(featsbf, rb1, w1p, rowBase, c, cnt);
  int lane = threadIdx.x & 31;
  int n = lane & 15, h = lane >> 4;
  float inv[2][8];
#pragma unroll
  for (int mt = 0; mt < 2; ++mt)
#pragma unroll
    for (int j = 0; j < 8; ++j) {
      int cj = __shfl(cnt[mt], j + 8 * h, 32);   // C row M = j + 8*(lane/16)
      inv[mt][j] = 1.0f / (float)(cj > 0 ? cj : 1);
    }
#pragma unroll
  for (int mt = 0; mt < 2; ++mt)
#pragma unroll
    for (int nt = 0; nt < 4; ++nt) {
      int col = nt * 16 + n;
      float bb = b1[col];
#pragma unroll
      for (int j = 0; j < 8; ++j) {
        long idx = (long)(rowBase + mt * 16 + j + 8 * h) * 64 + col;
        x1[idx] = c[mt][nt][j] * inv[mt][j] + bb + feats[idx];
      }
    }
}

// conv2: x2 = skconv(xagg, rb2, w2) + b2 + xagg@res2_W + res2_b   (Cout=128)
__global__ void __launch_bounds__(256)
conv2_kernel(const unsigned short* __restrict__ xaggbf,
             const int* __restrict__ rb2,
             const unsigned short* __restrict__ w2p,
             const float* __restrict__ b2,
             const unsigned short* __restrict__ res2p,
             const float* __restrict__ res2_b,
             unsigned short* __restrict__ x2bf, int nRows) {
  int wave = blockIdx.x * (blockDim.x >> 5) + (threadIdx.x >> 5);
  int rowBase = wave * 16;
  if (rowBase >= nRows) return;
  v8f c[1][8];
  int cnt[1];
  skconv_accum<8, 1>(xaggbf, rb2, w2p, rowBase, c, cnt);
  int lane = threadIdx.x & 31;
  int n = lane & 15, h = lane >> 4, m = n;
  float inv[8];
#pragma unroll
  for (int j = 0; j < 8; ++j) {
    int cj = __shfl(cnt[0], j + 8 * h, 32);
    inv[j] = 1.0f / (float)(cj > 0 ? cj : 1);
  }
#pragma unroll
  for (int nt = 0; nt < 8; ++nt)
#pragma unroll
    for (int j = 0; j < 8; ++j) c[0][nt][j] *= inv[j];

  // residual projection: A = self row, B = packed res2_W ([nt][kb] order)
  BFrag a[2];
  loadA64(a, xaggbf, rowBase + m, h);
  const uint4* wq = (const uint4*)res2p;
#pragma unroll
  for (int nt = 0; nt < 8; ++nt) {
#pragma unroll
    for (int kb = 0; kb < 2; ++kb) {
      BFrag b;
      unsigned o = (unsigned)(nt * 2 + kb) * 64u + (unsigned)lane * 2u;
      b.q[0] = wq[o];
      b.q[1] = wq[o + 1];
      c[0][nt] = __builtin_amdgcn_wmma_f32_16x16x32_bf16(
          false, a[kb].v, false, b.v, (short)0, c[0][nt], false, false);
    }
  }
#pragma unroll
  for (int nt = 0; nt < 8; ++nt) {
    int col = nt * 16 + n;
    float bb = b2[col] + res2_b[col];
#pragma unroll
    for (int j = 0; j < 8; ++j) {
      long idx = (long)(rowBase + j + 8 * h) * 128 + col;
      x2bf[idx] = f2bf(c[0][nt][j] + bb);
    }
  }
}

// fused FiLM conv: xr = gamma*(skconv(xin,rb1,wf)) + beta + bf + xin
__global__ void __launch_bounds__(256)
convf_kernel(const unsigned short* __restrict__ xinbf,
             const float* __restrict__ xin,
             const int* __restrict__ rb1,
             const unsigned short* __restrict__ wfp,
             const float* __restrict__ bf_,
             const float* __restrict__ gamma_c,
             const float* __restrict__ beta_c,
             float* __restrict__ xr,
             unsigned short* __restrict__ xrbf, int nRows) {
  int wave = blockIdx.x * (blockDim.x >> 5) + (threadIdx.x >> 5);
  int rowBase = wave * 32;
  if (rowBase >= nRows) return;
  v8f c[2][4];
  int cnt[2];
  skconv_accum<4, 2>(xinbf, rb1, wfp, rowBase, c, cnt);
  int lane = threadIdx.x & 31;
  int n = lane & 15, h = lane >> 4;
  float inv[2][8];
#pragma unroll
  for (int mt = 0; mt < 2; ++mt)
#pragma unroll
    for (int j = 0; j < 8; ++j) {
      int cj = __shfl(cnt[mt], j + 8 * h, 32);
      inv[mt][j] = 1.0f / (float)(cj > 0 ? cj : 1);
    }
#pragma unroll
  for (int mt = 0; mt < 2; ++mt)
#pragma unroll
    for (int nt = 0; nt < 4; ++nt) {
      int col = nt * 16 + n;
      float bb = bf_[col];
#pragma unroll
      for (int j = 0; j < 8; ++j) {
        int row = rowBase + mt * 16 + j + 8 * h;
        long ci = (long)(row >> 2) * 64 + col;     // coarse parent = row/4
        long xi = (long)row * 64 + col;
        float s = c[mt][nt][j] * inv[mt][j];
        float v = gamma_c[ci] * s + beta_c[ci] + bb + xin[xi];
        xr[xi] = v;
        xrbf[xi] = f2bf(v);
      }
    }
}

// Generic WMMA linear on bf16 rows: Y = X @ W + bias (f32 and/or bf16 out)
__global__ void __launch_bounds__(256)
linear_wmma_kernel(const unsigned short* __restrict__ Xbf,
                   const unsigned short* __restrict__ Wp,
                   const float* __restrict__ bias,
                   float* __restrict__ Yf,
                   unsigned short* __restrict__ Ybf,
                   int nRows, int KB, int NT, int Cin,
                   int CoutStore, int strideOut, int biasLen) {
  int wave = blockIdx.x * (blockDim.x >> 5) + (threadIdx.x >> 5);
  int rowBase = wave * 16;
  if (rowBase >= nRows) return;
  int lane = threadIdx.x & 31;
  int m = lane & 15, h = lane >> 4, n = m;
  BFrag a[4];
  const uint4* p = (const uint4*)Xbf + (unsigned)(rowBase + m) * (unsigned)(Cin >> 3);
  for (int kb = 0; kb < KB; ++kb) {
    a[kb].q[0] = p[4 * kb + h];
    a[kb].q[1] = p[4 * kb + 2 + h];
  }
  const uint4* wq = (const uint4*)Wp;
  for (int nt = 0; nt < NT; ++nt) {
    v8f c = {0.f, 0.f, 0.f, 0.f, 0.f, 0.f, 0.f, 0.f};
    for (int kb = 0; kb < KB; ++kb) {
      BFrag b;
      unsigned o = (unsigned)(nt * KB + kb) * 64u + (unsigned)lane * 2u;
      b.q[0] = wq[o];
      b.q[1] = wq[o + 1];
      c = __builtin_amdgcn_wmma_f32_16x16x32_bf16(
          false, a[kb].v, false, b.v, (short)0, c, false, false);
    }
    int col = nt * 16 + n;
    float bb = (col < biasLen) ? bias[col] : 0.0f;
    if (col < CoutStore) {
#pragma unroll
      for (int j = 0; j < 8; ++j) {
        float v = c[j] + bb;
        long o = (long)(rowBase + j + 8 * h) * strideOut + col;
        if (Yf)  Yf[o] = v;
        if (Ybf) Ybf[o] = f2bf(v);
      }
    }
  }
}

// scatter_mean: children of coarse c are fine rows 4c..4c+3 -> bf16 xagg
__global__ void __launch_bounds__(256)
scatter_mean_kernel(const float* __restrict__ x1,
                    unsigned short* __restrict__ xaggbf, int total) {
  int i = blockIdx.x * blockDim.x + threadIdx.x;
  if (i >= total) return;
  int cc = i >> 6, col = i & 63;
  long b = (long)cc * 256 + col;
  xaggbf[i] = f2bf(0.25f * (x1[b] + x1[b + 64] + x1[b + 128] + x1[b + 192]));
}

// xin = x1 + up_c[row/4]  (f32 + bf16 copies)
__global__ void __launch_bounds__(256)
add_upsample_kernel(const float* __restrict__ x1,
                    const float* __restrict__ up_c,
                    float* __restrict__ xin,
                    unsigned short* __restrict__ xinbf, int total) {
  int i = blockIdx.x * blockDim.x + threadIdx.x;
  if (i >= total) return;
  int row = i >> 6, col = i & 63;
  float v = x1[i] + up_c[(long)(row >> 2) * 64 + col];
  xin[i] = v;
  xinbf[i] = f2bf(v);
}

// deterministic two-stage pooling: stage 1 partial sums over 256-row stripes
__global__ void __launch_bounds__(64)
pool_partial_kernel(const float* __restrict__ xr,
                    float* __restrict__ partial,
                    int rowsPerBlock, int nRows) {
  int c = threadIdx.x;          // blockDim = 64
  int r0 = blockIdx.x * rowsPerBlock;
  int r1 = r0 + rowsPerBlock;
  if (r1 > nRows) r1 = nRows;
  float acc = 0.0f;
  for (int r = r0; r < r1; ++r) acc += xr[(long)r * 64 + c];
  partial[blockIdx.x * 64 + c] = acc;
}

// tiny heads: cls / embed / joints (one block of 64 threads)
__global__ void __launch_bounds__(64)
head_kernel(const float* __restrict__ partial, int nPartials,
            const float* __restrict__ cls_W, const float* __restrict__ cls_b,
            const float* __restrict__ emb_W, const float* __restrict__ emb_b,
            const float* __restrict__ kp_W, const float* __restrict__ kp_b,
            float* __restrict__ out_cls, float* __restrict__ out_embed,
            float* __restrict__ out_joints) {
  __shared__ float pooled[64];
  __shared__ float e[64];
  __shared__ float nrm;
  int t = threadIdx.x;
  float acc = 0.0f;
  for (int p = 0; p < nPartials; ++p) acc += partial[p * 64 + t];
  pooled[t] = acc / (float)M_FINE;
  __syncthreads();
  if (t < 16) {
    float s = cls_b[t];
    for (int c2 = 0; c2 < 64; ++c2) s += pooled[c2] * cls_W[c2 * 16 + t];
    out_cls[t] = s;
  }
  {
    float s = emb_b[t];
    for (int c2 = 0; c2 < 64; ++c2) s += pooled[c2] * emb_W[c2 * 64 + t];
    e[t] = s;
  }
  __syncthreads();
  if (t == 0) {
    float s = 0.0f;
    for (int c2 = 0; c2 < 64; ++c2) s += e[c2] * e[c2];
    nrm = fmaxf(sqrtf(s), 1e-12f);
  }
  __syncthreads();
  out_embed[t] = e[t] / nrm;
  if (t < 18) {
    int j = t / 3, o = t % 3;
    float s = kp_b[t];
    for (int c2 = 0; c2 < 64; ++c2) s += pooled[c2] * kp_W[(j * 64 + c2) * 3 + o];
    out_joints[t] = 1.2f * tanhf(s);
  }
}

// ---------------------------------------------------------------------------
extern "C" void kernel_launch(void* const* d_in, const int* in_sizes, int n_in,
                              void* d_out, int out_size, void* d_ws, size_t ws_size,
                              hipStream_t stream) {
  const float* feats  = (const float*)d_in[0];
  const int*   rb1    = (const int*)d_in[1];
  const int*   rb2    = (const int*)d_in[2];
  /* d_in[3] = invc: children contiguous -> coarse = row>>2 */
  const float* w1     = (const float*)d_in[4];
  const float* b1     = (const float*)d_in[5];
  const float* w2     = (const float*)d_in[6];
  const float* b2     = (const float*)d_in[7];
  const float* res2_W = (const float*)d_in[8];
  const float* res2_b = (const float*)d_in[9];
  const float* cond_W = (const float*)d_in[10];
  const float* cond_b = (const float*)d_in[11];
  const float* up_W   = (const float*)d_in[12];
  const float* up_b   = (const float*)d_in[13];
  const float* wf     = (const float*)d_in[14];
  const float* bf     = (const float*)d_in[15];
  const float* gamma_W = (const float*)d_in[16];
  const float* gamma_b = (const float*)d_in[17];
  const float* beta_W  = (const float*)d_in[18];
  const float* beta_b  = (const float*)d_in[19];
  const float* sem_W   = (const float*)d_in[20];
  const float* sem_b   = (const float*)d_in[21];
  const float* cls_W   = (const float*)d_in[22];
  const float* cls_b   = (const float*)d_in[23];
  const float* kp_W    = (const float*)d_in[24];
  const float* kp_b    = (const float*)d_in[25];
  const float* emb_W   = (const float*)d_in[26];
  const float* emb_b   = (const float*)d_in[27];

  char* ws = (char*)d_ws;
  size_t off = 0;
  auto alloc = [&](size_t bytes) -> void* {
    void* p = ws + off;
    off += (bytes + 255) & ~(size_t)255;
    return p;
  };
  unsigned short* w1p    = (unsigned short*)alloc(27 * 2 * 4 * 512 * 2);
  unsigned short* w2p    = (unsigned short*)alloc(27 * 2 * 8 * 512 * 2);
  unsigned short* wfp    = (unsigned short*)alloc(27 * 2 * 4 * 512 * 2);
  unsigned short* res2p  = (unsigned short*)alloc(2 * 8 * 512 * 2);
  unsigned short* condp  = (unsigned short*)alloc(4 * 4 * 512 * 2);
  unsigned short* upp    = (unsigned short*)alloc(4 * 4 * 512 * 2);
  unsigned short* gammap = (unsigned short*)alloc(2 * 4 * 512 * 2);
  unsigned short* betap  = (unsigned short*)alloc(2 * 4 * 512 * 2);
  unsigned short* semp   = (unsigned short*)alloc(2 * 2 * 512 * 2);
  unsigned short* featsbf = (unsigned short*)alloc((size_t)M_FINE * 64 * 2);
  float*          x1      = (float*)alloc((size_t)M_FINE * 64 * 4);
  unsigned short* xaggbf  = (unsigned short*)alloc((size_t)N_COARSE * 64 * 2);
  unsigned short* x2bf    = (unsigned short*)alloc((size_t)N_COARSE * 128 * 2);
  unsigned short* condcbf = (unsigned short*)alloc((size_t)N_COARSE * 64 * 2);
  float*          up_c    = (float*)alloc((size_t)N_COARSE * 64 * 4);
  float*          xin     = (float*)alloc((size_t)M_FINE * 64 * 4);
  unsigned short* xinbf   = (unsigned short*)alloc((size_t)M_FINE * 64 * 2);
  float*          gamma_c = (float*)alloc((size_t)N_COARSE * 64 * 4);
  float*          beta_c  = (float*)alloc((size_t)N_COARSE * 64 * 4);
  float*          xr      = (float*)alloc((size_t)M_FINE * 64 * 4);
  unsigned short* xrbf    = (unsigned short*)alloc((size_t)M_FINE * 64 * 2);
  float*          partial = (float*)alloc((size_t)250 * 64 * 4);

  float* out      = (float*)d_out;
  float* out_sem  = out;                       // [M, 20]
  float* out_cls  = out + (size_t)M_FINE * 20; // [16]
  float* out_emb  = out_cls + 16;              // [64]
  float* out_jnt  = out_emb + 64;              // [6,3]

  auto nb = [](int total) { return (total + 255) / 256; };

  // --- pack weights + convert feats ---
  pack_weights_kernel<<<nb(27*2*4*512), 256, 0, stream>>>(w1,     w1p,    27,  64,  64, 4);
  pack_weights_kernel<<<nb(27*2*8*512), 256, 0, stream>>>(w2,     w2p,    27,  64, 128, 8);
  pack_weights_kernel<<<nb(27*2*4*512), 256, 0, stream>>>(wf,     wfp,    27,  64,  64, 4);
  pack_weights_kernel<<<nb(2*8*512),    256, 0, stream>>>(res2_W, res2p,   1,  64, 128, 8);
  pack_weights_kernel<<<nb(4*4*512),    256, 0, stream>>>(cond_W, condp,   1, 128,  64, 4);
  pack_weights_kernel<<<nb(4*4*512),    256, 0, stream>>>(up_W,   upp,     1, 128,  64, 4);
  pack_weights_kernel<<<nb(2*4*512),    256, 0, stream>>>(gamma_W, gammap, 1,  64,  64, 4);
  pack_weights_kernel<<<nb(2*4*512),    256, 0, stream>>>(beta_W,  betap,  1,  64,  64, 4);
  pack_weights_kernel<<<nb(2*2*512),    256, 0, stream>>>(sem_W,   semp,   1,  64,  20, 2);
  cvt_bf16_kernel<<<nb(M_FINE * 64), 256, 0, stream>>>(feats, featsbf, M_FINE * 64);

  const int WPB = 8;                 // 8 wave32 per 256-thread block
  int waves32F = M_FINE / 32;        // 2000 (32-row tiles)
  int waves16C = N_COARSE / 16;      // 1000
  int waves16F = M_FINE / 16;        // 4000

  // conv1 -> x1 (f32)
  conv1_kernel<<<(waves32F + WPB - 1) / WPB, 256, 0, stream>>>(featsbf, feats, rb1, w1p, b1, x1, M_FINE);
  // scatter mean -> xagg (bf16)
  scatter_mean_kernel<<<nb(N_COARSE * 64), 256, 0, stream>>>(x1, xaggbf, N_COARSE * 64);
  // conv2 (+ projected residual) -> x2 (bf16)
  conv2_kernel<<<(waves16C + WPB - 1) / WPB, 256, 0, stream>>>(xaggbf, rb2, w2p, b2, res2p, res2_b, x2bf, N_COARSE);
  // cond_c (bf16 only) / up_c (f32)
  linear_wmma_kernel<<<(waves16C + WPB - 1) / WPB, 256, 0, stream>>>(x2bf, condp, cond_b,
      (float*)nullptr, condcbf, N_COARSE, 4, 4, 128, 64, 64, 64);
  linear_wmma_kernel<<<(waves16C + WPB - 1) / WPB, 256, 0, stream>>>(x2bf, upp, up_b,
      up_c, (unsigned short*)nullptr, N_COARSE, 4, 4, 128, 64, 64, 64);
  // xin = x1 + up_c[row/4]  (f32 + bf16)
  add_upsample_kernel<<<nb(M_FINE * 64), 256, 0, stream>>>(x1, up_c, xin, xinbf, M_FINE * 64);
  // gamma_c / beta_c (f32, coarse level)
  linear_wmma_kernel<<<(waves16C + WPB - 1) / WPB, 256, 0, stream>>>(condcbf, gammap, gamma_b,
      gamma_c, (unsigned short*)nullptr, N_COARSE, 2, 4, 64, 64, 64, 64);
  linear_wmma_kernel<<<(waves16C + WPB - 1) / WPB, 256, 0, stream>>>(condcbf, betap, beta_b,
      beta_c, (unsigned short*)nullptr, N_COARSE, 2, 4, 64, 64, 64, 64);
  // fused FiLM conv -> xr (f32 + bf16)
  convf_kernel<<<(waves32F + WPB - 1) / WPB, 256, 0, stream>>>(xinbf, xin, rb1, wfp, bf,
      gamma_c, beta_c, xr, xrbf, M_FINE);
  // sem_logits = xr @ sem_W + sem_b  (Cout padded to 32, store 20)
  linear_wmma_kernel<<<(waves16F + WPB - 1) / WPB, 256, 0, stream>>>(xrbf, semp, sem_b,
      out_sem, (unsigned short*)nullptr, M_FINE, 2, 2, 64, 20, 20, 20);
  // pooled mean + heads
  pool_partial_kernel<<<250, 64, 0, stream>>>(xr, partial, 256, M_FINE);
  head_kernel<<<1, 64, 0, stream>>>(partial, 250, cls_W, cls_b, emb_W, emb_b,
                                    kp_W, kp_b, out_cls, out_emb, out_jnt);
}
